// BinaryConv2d_22471268892738
// MI455X (gfx1250) — compile-verified, matrix operand order checked by reading, not run on hardware
//
#include <hip/hip_runtime.h>
#include <hip/hip_bf16.h>

typedef _Float16 v8h  __attribute__((ext_vector_type(8)));
typedef _Float16 v16h __attribute__((ext_vector_type(16)));
typedef float    v8f  __attribute__((ext_vector_type(8)));

#define BATCH 32
#define CIN   64
#define COUT  64
#define HH    224
#define WW    224

// LDS geometry
#define WK    192   // k-extent per kh block: kw*64 + c
#define WKP   200   // padded k stride (200*2B = 400B, multiple of 16)
#define XWP   226   // input w positions (w=-1..224 -> wp=0..225)
#define XCP   72    // padded channel stride (72*2B = 144B, multiple of 16)

__device__ __forceinline__ v16h load_frag(const _Float16* p) {
    // Two 16B LDS loads; halves j=0..7 = K base..base+7, j=8..15 = K base+16..+23
    v8h lo = *(const v8h*)(p);
    v8h hi = *(const v8h*)(p + 16);
    v16h r;
    #pragma unroll
    for (int i = 0; i < 8; ++i) { r[i] = lo[i]; r[i + 8] = hi[i]; }
    return r;
}

// B-fragment LDS address for linear step t = ks*7 + nt
__device__ __forceinline__ const _Float16* bptr(const _Float16* xBase, int t) {
    const int ks = t / 7;
    const int nt = t - ks * 7;
    const int kw = ks >> 1;
    const int c0 = (ks & 1) * 32;
    return xBase + (nt * 16 + kw) * XCP + c0;
}

__global__ __launch_bounds__(256)
void bin_conv3x3_wmma(const float* __restrict__ x,
                      const float* __restrict__ weight,
                      const float* __restrict__ bias,
                      float* __restrict__ out)
{
    __shared__ __attribute__((aligned(16))) _Float16 lds_w[COUT * WKP]; // 25,600 B
    __shared__ __attribute__((aligned(16))) _Float16 lds_x[XWP * XCP];  // 32,544 B

    const int tid  = threadIdx.x;
    const int lane = tid & 31;
    const int wave = tid >> 5;

    const int bh = blockIdx.x;
    const int b  = bh / HH;
    const int h  = bh % HH;

    const int mBase = (wave & 3) * 16;     // 4 M-tiles of 16 (COUT)
    const int wBase = (wave >> 2) * 112;   // 2 N-halves of 112 pixels

    const int lmod  = lane & 15;           // N (for B/C/D) or M (for A)
    const int lhalf = (lane < 16) ? 0 : 8; // K offset for upper half-wave

    v8f acc[7] = {};

    // Fragment base pointers (per-wave constants)
    const _Float16* aBase = &lds_w[(mBase + lmod) * WKP + lhalf];
    const _Float16* xBase = &lds_x[(wBase + lmod) * XCP + lhalf];

    for (int kh = 0; kh < 3; ++kh) {
        __syncthreads();

        // ---- stage binarized weights for this kh: lds_w[o*WKP + kw*64 + c]
        for (int idx = tid; idx < COUT * WK; idx += 256) {
            int o  = idx / WK;
            int r  = idx - o * WK;
            int kw = r >> 6;
            int c  = r & 63;
            float v = weight[((o * CIN + c) * 3 + kh) * 3 + kw];
            float s = (v > 0.f) ? 1.f : ((v < 0.f) ? -1.f : 0.f);
            lds_w[o * WKP + r] = (_Float16)s;
        }

        // ---- stage input row h+kh-1 transposed to [wp][c] f16 (zero halo)
        const int  hrow  = h + kh - 1;
        const bool rowOK = (hrow >= 0) && (hrow < HH);   // block-uniform
        if (rowOK) {
            // interior: float4 loads along w (16B aligned: row base % 16 == 0)
            for (int idx = tid; idx < CIN * (WW / 4); idx += 256) {
                int c = idx / (WW / 4);
                int w = (idx - c * (WW / 4)) * 4;
                const float4 v = *(const float4*)(&x[((b * CIN + c) * HH + hrow) * WW + w]);
                _Float16* dst = &lds_x[(w + 1) * XCP + c];
                dst[0 * XCP] = (_Float16)v.x;
                dst[1 * XCP] = (_Float16)v.y;
                dst[2 * XCP] = (_Float16)v.z;
                dst[3 * XCP] = (_Float16)v.w;
            }
            // halo columns wp=0 (w=-1) and wp=225 (w=224) are always zero padding
            if (tid < 128) {
                int c    = tid & 63;
                int side = tid >> 6;
                lds_x[(side ? (XWP - 1) : 0) * XCP + c] = (_Float16)0.f;
            }
        } else {
            for (int idx = tid; idx < CIN * XWP; idx += 256) {
                int c  = idx / XWP;
                int wp = idx - c * XWP;
                lds_x[wp * XCP + c] = (_Float16)0.f;
            }
        }

        // ---- prefetch next row into L2 while we compute on this one
        if (kh < 2) {
            int hn = h + kh;  // row needed by next kh iteration
            if (hn < HH) {
                for (int idx = tid; idx < CIN * (WW / 4); idx += 256 * 4) {
                    int c = idx / (WW / 4);
                    int w = (idx - c * (WW / 4)) * 4;
                    __builtin_prefetch(&x[((b * CIN + c) * HH + hn) * WW + w], 0, 0);
                }
            }
        }

        __syncthreads();

        // ---- pinned-schedule WMMA sweep: 42 steps (6 K-steps x 7 N-subtiles),
        //      2-deep B lookahead, A prefetched 3 steps before each K-step switch.
        v16h a_cur = load_frag(aBase);            // A(ks=0)
        v16h b0    = load_frag(bptr(xBase, 0));   // B(t=0)
        v16h b1    = load_frag(bptr(xBase, 1));   // B(t=1)
        __builtin_amdgcn_sched_group_barrier(0x100, 6, 0); // prime: 6 DS reads
        v16h a_nxt = a_cur;

        #pragma unroll
        for (int t = 0; t < 42; ++t) {
            const int  ks     = t / 7;
            const int  nt     = t - ks * 7;
            const bool haveB2 = (t + 2 < 42);
            const bool loadA  = (nt == 4) && (ks < 5);

            v16h b2 = b1;
            if (haveB2) b2 = load_frag(bptr(xBase, t + 2));
            if (loadA)  a_nxt = load_frag(aBase + ((ks + 1) >> 1) * 64 +
                                                  ((ks + 1) & 1) * 32);

            acc[nt] = __builtin_amdgcn_wmma_f32_16x16x32_f16(
                false, a_cur, false, b0, (short)0, acc[nt], false, false);

            if (nt == 6) a_cur = a_nxt;
            b0 = b1;
            b1 = b2;

            // scheduling spec: [B(t+2) loads (+A-next loads)] then [WMMA(t)]
            if (haveB2 && loadA) {
                __builtin_amdgcn_sched_group_barrier(0x100, 4, 0);
            } else if (haveB2 || loadA) {
                __builtin_amdgcn_sched_group_barrier(0x100, 2, 0);
            }
            __builtin_amdgcn_sched_group_barrier(0x008, 1, 0);
        }
    }

    // ---- epilogue: C/D layout is lane%16 = N (pixel), VGPR r (+8 upper) = M (cout)
    #pragma unroll
    for (int r = 0; r < 8; ++r) {
        const int o  = mBase + r + lhalf;
        const float bv = bias[o];
        #pragma unroll
        for (int nt = 0; nt < 7; ++nt) {
            const int w = wBase + nt * 16 + lmod;
            out[((b * COUT + o) * HH + h) * WW + w] = acc[nt][r] + bv;
        }
    }
}

extern "C" void kernel_launch(void* const* d_in, const int* in_sizes, int n_in,
                              void* d_out, int out_size, void* d_ws, size_t ws_size,
                              hipStream_t stream) {
    const float* x      = (const float*)d_in[0];
    const float* weight = (const float*)d_in[1];
    const float* bias   = (const float*)d_in[2];
    float* out          = (float*)d_out;

    dim3 grid(BATCH * HH);   // one (b, h) output row per workgroup
    dim3 block(256);         // 8 wave32 waves
    bin_conv3x3_wmma<<<grid, block, 0, stream>>>(x, weight, bias, out);
}
